// MultiHeadAttention_16174846837067
// MI455X (gfx1250) — matrix operand order repaired and checked by reading, not compiled, not run on hardware
//
#include <hip/hip_runtime.h>
#include <hip/hip_bf16.h>

typedef __attribute__((ext_vector_type(16))) __bf16 v16bf;
typedef __attribute__((ext_vector_type(8)))  __bf16 v8bf;
typedef __attribute__((ext_vector_type(4)))  __bf16 v4bf;
typedef __attribute__((ext_vector_type(8)))  float  v8f;
typedef __attribute__((ext_vector_type(4)))  float  v4f;
typedef __attribute__((ext_vector_type(4)))  int    v4i;

union BF16x16 { v16bf v; v8bf h[2]; };

static constexpr int D_MODEL = 1024;
static constexpr int SEQ     = 2048;

// CDNA5 async global->LDS copy (ASYNCcnt path), if this toolchain declares it.
#if defined(__AMDGCN__) && __has_builtin(__builtin_amdgcn_global_load_async_to_lds_b128)
#define HAVE_ASYNC_LDS 1
#else
#define HAVE_ASYNC_LDS 0
#endif

#if HAVE_ASYNC_LDS
typedef __attribute__((address_space(1))) v4i g_v4i;
typedef __attribute__((address_space(3))) v4i l_v4i;
__device__ inline void async_g2l_b128(const __bf16* g, __bf16* l) {
  __builtin_amdgcn_global_load_async_to_lds_b128(
      (g_v4i*)(void*)g, (l_v4i*)(void*)l, 0, 0);
}
__device__ inline void wait_async0() {
#if __has_builtin(__builtin_amdgcn_s_wait_asynccnt)
  __builtin_amdgcn_s_wait_asynccnt(0);
#else
  asm volatile("s_wait_asynccnt 0" ::: "memory");
#endif
}
#endif

__device__ inline __bf16 f2bf(float f) {
  unsigned u = __builtin_bit_cast(unsigned, f);
  u += 0x7FFFu + ((u >> 16) & 1u);            // round-to-nearest-even
  unsigned short h = (unsigned short)(u >> 16);
  return __builtin_bit_cast(__bf16, h);
}

// ---------------------------------------------------------------------------
// Generic bf16 WMMA GEMM:  Out[M,N] = A[M,K] * W[K,N] + bias
// A_BF16:   0 -> A is fp32 (converted to bf16 while staging), 1 -> A already bf16
// OUT_MODE: 0 -> fp32 row-major, 1 -> bf16 row-major, 2 -> bf16 transposed [b][n][t]
// ---------------------------------------------------------------------------
template<int A_BF16, int OUT_MODE>
__global__ __launch_bounds__(256)
void wmma_gemm(const void* __restrict__ Aptr, const float* __restrict__ Bw,
               const float* __restrict__ bias, void* __restrict__ Out,
               int M, int N, int K)
{
  constexpr int BM = 128, BN = 128, BK = 32;
  constexpr int ASTR = BK + 16;   // 48 elems -> 96B rows (16B aligned)
  constexpr int BSTR = BK + 16;
  __shared__ __bf16 As[BM * ASTR];
  __shared__ __bf16 Bts[BN * BSTR];   // transposed: Bts[n][k]

  const int tid  = threadIdx.x;
  const int lane = tid & 31;
  const int w    = tid >> 5;          // 8 waves
  const int wm   = w & 1;             // 2 waves in M
  const int wn   = w >> 1;            // 4 waves in N
  const int hh   = lane >> 4;         // half-wave select
  const int l16  = lane & 15;
  const int m0   = blockIdx.y * BM;
  const int n0   = blockIdx.x * BN;

  v8f zero8 = {0.f,0.f,0.f,0.f,0.f,0.f,0.f,0.f};
  v8f acc[4][2];
  #pragma unroll
  for (int i = 0; i < 4; ++i)
    #pragma unroll
    for (int j = 0; j < 2; ++j) acc[i][j] = zero8;

  const float*  Af = (const float*)Aptr;
  const __bf16* Ab = (const __bf16*)Aptr;

  for (int k0 = 0; k0 < K; k0 += BK) {
    __syncthreads();
    // ---- stage A tile (BMxBK) into LDS as bf16, row-major
    if (A_BF16) {
#if HAVE_ASYNC_LDS
      #pragma unroll
      for (int i = 0; i < 2; ++i) {
        int idx = tid + 256 * i;
        int row = idx >> 2, c8 = (idx & 3) * 8;
        async_g2l_b128(Ab + (size_t)(m0 + row) * K + k0 + c8,
                       &As[row * ASTR + c8]);
      }
#else
      #pragma unroll
      for (int i = 0; i < 2; ++i) {
        int idx = tid + 256 * i;
        int row = idx >> 2, c8 = (idx & 3) * 8;
        v8bf d = *(const v8bf*)(Ab + (size_t)(m0 + row) * K + k0 + c8);
        *(v8bf*)(&As[row * ASTR + c8]) = d;
      }
#endif
    } else {
      #pragma unroll
      for (int i = 0; i < 4; ++i) {
        int idx = tid + 256 * i;
        int row = idx >> 3, c4 = (idx & 7) * 4;
        v4f d = *(const v4f*)(Af + (size_t)(m0 + row) * K + k0 + c4);
        v4bf hb;
        #pragma unroll
        for (int j = 0; j < 4; ++j) hb[j] = f2bf(d[j]);
        *(v4bf*)(&As[row * ASTR + c4]) = hb;
      }
    }
    // ---- stage W tile (BKxBN) transposed into LDS: Bts[n][k]
    #pragma unroll
    for (int i = 0; i < 4; ++i) {
      int idx = tid + 256 * i;
      int kk = idx >> 5, n4 = (idx & 31) * 4;
      v4f d = *(const v4f*)(Bw + (size_t)(k0 + kk) * N + n0 + n4);
      #pragma unroll
      for (int j = 0; j < 4; ++j) Bts[(n4 + j) * BSTR + kk] = f2bf(d[j]);
    }
#if HAVE_ASYNC_LDS
    if (A_BF16) wait_async0();
#endif
    __syncthreads();

    // ---- fragments + WMMA (one K-step of 32 per staged tile)
    BF16x16 afr[4], bfr[2];
    #pragma unroll
    for (int mt = 0; mt < 4; ++mt) {
      int row = wm * 64 + mt * 16 + l16;
      const __bf16* p = &As[row * ASTR + hh * 16];
      afr[mt].h[0] = *(const v8bf*)p;
      afr[mt].h[1] = *(const v8bf*)(p + 8);
    }
    #pragma unroll
    for (int nt = 0; nt < 2; ++nt) {
      int col = wn * 32 + nt * 16 + l16;
      const __bf16* p = &Bts[col * BSTR + hh * 16];
      bfr[nt].h[0] = *(const v8bf*)p;
      bfr[nt].h[1] = *(const v8bf*)(p + 8);
    }
    #pragma unroll
    for (int mt = 0; mt < 4; ++mt)
      #pragma unroll
      for (int nt = 0; nt < 2; ++nt)
        acc[mt][nt] = __builtin_amdgcn_wmma_f32_16x16x32_bf16(
            false, afr[mt].v, false, bfr[nt].v, (short)0, acc[mt][nt], false, false);
  }

  // ---- epilogue: bias + store
  #pragma unroll
  for (int nt = 0; nt < 2; ++nt) {
    int n = n0 + wn * 32 + nt * 16 + l16;
    float bv = bias[n];
    #pragma unroll
    for (int mt = 0; mt < 4; ++mt) {
      #pragma unroll
      for (int j = 0; j < 8; ++j) {
        int m = m0 + wm * 64 + mt * 16 + j + 8 * hh;
        float vv = acc[mt][nt][j] + bv;
        if (OUT_MODE == 0) {
          ((float*)Out)[(size_t)m * N + n] = vv;
        } else if (OUT_MODE == 1) {
          ((__bf16*)Out)[(size_t)m * N + n] = f2bf(vv);
        } else {
          int bb = m >> 11, t = m & (SEQ - 1);        // SEQ = 2048
          ((__bf16*)Out)[((size_t)bb * N + n) * SEQ + t] = f2bf(vv);
        }
      }
    }
  }
}

// ---------------------------------------------------------------------------
// Flash attention (causal, full d_model head).  One 16-row Q tile per block.
// Q,K: bf16 [B*T, C] row-major.  Vt: bf16 [B][C][T] (transposed).  Out: bf16.
// 8 waves: wave w computes score cols [s0+16w,+16) and owns output cols
// [128w, +128).
// ---------------------------------------------------------------------------
__global__ __launch_bounds__(256)
void flash_attn(const __bf16* __restrict__ Q, const __bf16* __restrict__ Kb,
                const __bf16* __restrict__ Vt, __bf16* __restrict__ Out)
{
  constexpr int C = D_MODEL, T = SEQ;
  constexpr int QSTR = C + 8;     // 1032 elems -> 2064B rows (16B aligned)
  constexpr int SSTR = 132;
  constexpr int PSTR = 144;       // 288B rows (16B aligned)
  __shared__ __bf16 Qs[16 * QSTR];
  __shared__ float  Ss[16 * SSTR];
  __shared__ __bf16 Ps[16 * PSTR];
  __shared__ float  redbuf[16 * 16];
  __shared__ float  row_m[16], row_l[16], row_alpha[16];

  const int tid  = threadIdx.x;
  const int lane = tid & 31;
  const int w    = tid >> 5;
  const int hh   = lane >> 4;
  const int l16  = lane & 15;

  const int bidx = blockIdx.x >> 7;       // T/16 = 128 q-tiles per batch
  const int qt   = blockIdx.x & 127;
  const int m0   = qt * 16;
  const size_t gr0 = (size_t)bidx * T + m0;

  // ---- stage 16x1024 Q tile into LDS (async copy when available)
  #pragma unroll
  for (int i = 0; i < 8; ++i) {
    int idx = tid + 256 * i;
    int row = idx >> 7, c8 = (idx & 127) * 8;
#if HAVE_ASYNC_LDS
    async_g2l_b128(Q + (gr0 + row) * C + c8, &Qs[row * QSTR + c8]);
#else
    *(v8bf*)(&Qs[row * QSTR + c8]) = *(const v8bf*)(Q + (gr0 + row) * C + c8);
#endif
  }
  if (tid < 16) { row_m[tid] = -1e30f; row_l[tid] = 0.0f; }
#if HAVE_ASYNC_LDS
  wait_async0();
#endif
  __syncthreads();

  v8f zero8 = {0.f,0.f,0.f,0.f,0.f,0.f,0.f,0.f};
  v8f oacc[8];
  #pragma unroll
  for (int i = 0; i < 8; ++i) oacc[i] = zero8;

  const int   cw0     = w * 128;
  const int   nchunks = m0 / 128 + 1;     // causal: only chunks with s <= m0+15
  const float scale   = 0.03125f;         // 1/sqrt(1024)

  for (int ch = 0; ch < nchunks; ++ch) {
    const int s0 = ch * 128;

    // ---- scores: S_w = Q(16x1024) @ K[s0+16w..+16)^T
    v8f sc = zero8;
    const int scol = s0 + w * 16 + l16;
    const __bf16* krow = Kb + ((size_t)bidx * T + scol) * C;
    #pragma unroll 4
    for (int kk = 0; kk < C; kk += 32) {
      BF16x16 a, b;
      const __bf16* qa = &Qs[l16 * QSTR + kk + hh * 16];
      a.h[0] = *(const v8bf*)qa;
      a.h[1] = *(const v8bf*)(qa + 8);
      const __bf16* kp = krow + kk + hh * 16;   // 16 contiguous K per lane, same k order as A
      b.h[0] = *(const v8bf*)kp;
      b.h[1] = *(const v8bf*)(kp + 8);
      sc = __builtin_amdgcn_wmma_f32_16x16x32_bf16(false, a.v, false, b.v,
                                                   (short)0, sc, false, false);
    }
    // causal mask + scale, write to LDS score buffer
    #pragma unroll
    for (int j = 0; j < 8; ++j) {
      int r = j + 8 * hh;
      float sv = (scol <= m0 + r) ? sc[j] * scale : -1e30f;
      Ss[r * SSTR + w * 16 + l16] = sv;
    }
    __syncthreads();

    // ---- online softmax stats (cooperative over 16 rows x 128 cols)
    {
      int r = tid & 15, cg = tid >> 4;
      float pm = -1e30f;
      #pragma unroll
      for (int i = 0; i < 8; ++i) pm = fmaxf(pm, Ss[r * SSTR + cg * 8 + i]);
      redbuf[r * 16 + cg] = pm;
    }
    __syncthreads();
    if (tid < 16) {
      int r = tid;
      float cm = -1e30f;
      for (int i = 0; i < 16; ++i) cm = fmaxf(cm, redbuf[r * 16 + i]);
      float mn = fmaxf(row_m[r], cm);
      row_alpha[r] = __expf(row_m[r] - mn);
      row_m[r] = mn;
    }
    __syncthreads();
    {
      int r = tid & 15, cg = tid >> 4;
      float mr = row_m[r];
      float ps = 0.0f;
      #pragma unroll
      for (int i = 0; i < 8; ++i) {
        int c = cg * 8 + i;
        float p = __expf(Ss[r * SSTR + c] - mr);
        Ps[r * PSTR + c] = f2bf(p);
        ps += p;
      }
      redbuf[r * 16 + cg] = ps;
    }
    __syncthreads();
    if (tid < 16) {
      int r = tid;
      float s = 0.0f;
      for (int i = 0; i < 16; ++i) s += redbuf[r * 16 + i];
      row_l[r] = row_alpha[r] * row_l[r] + s;
    }
    __syncthreads();

    // ---- rescale running output, then O += P(16x128) @ V_chunk(128 x 128-slice)
    float al[8];
    #pragma unroll
    for (int j = 0; j < 8; ++j) al[j] = row_alpha[j + 8 * hh];
    #pragma unroll
    for (int nt = 0; nt < 8; ++nt)
      #pragma unroll
      for (int j = 0; j < 8; ++j) oacc[nt][j] *= al[j];

    #pragma unroll
    for (int kk = 0; kk < 128; kk += 32) {
      BF16x16 a;
      const __bf16* pa = &Ps[l16 * PSTR + kk + hh * 16];
      a.h[0] = *(const v8bf*)pa;
      a.h[1] = *(const v8bf*)(pa + 8);
      #pragma unroll
      for (int nt = 0; nt < 8; ++nt) {
        int c = cw0 + nt * 16 + l16;
        const __bf16* vb = Vt + ((size_t)bidx * C + c) * T + s0 + kk + hh * 16;
        BF16x16 b;
        b.h[0] = *(const v8bf*)vb;
        b.h[1] = *(const v8bf*)(vb + 8);
        oacc[nt] = __builtin_amdgcn_wmma_f32_16x16x32_bf16(
            false, a.v, false, b.v, (short)0, oacc[nt], false, false);
      }
    }
    __syncthreads();
  }

  // ---- finalize: divide by row sums, store bf16
  float il[8];
  #pragma unroll
  for (int j = 0; j < 8; ++j) il[j] = 1.0f / row_l[j + 8 * hh];
  #pragma unroll
  for (int nt = 0; nt < 8; ++nt) {
    int c = cw0 + nt * 16 + l16;
    #pragma unroll
    for (int j = 0; j < 8; ++j) {
      int r = j + 8 * hh;
      Out[(gr0 + r) * C + c] = f2bf(oacc[nt][j] * il[j]);
    }
  }
}

// ---------------------------------------------------------------------------
extern "C" void kernel_launch(void* const* d_in, const int* in_sizes, int n_in,
                              void* d_out, int out_size, void* d_ws, size_t ws_size,
                              hipStream_t stream) {
  (void)in_sizes; (void)n_in; (void)out_size; (void)ws_size;
  const float* x  = (const float*)d_in[0];
  const float* Wq = (const float*)d_in[1];
  const float* bq = (const float*)d_in[2];
  const float* Wk = (const float*)d_in[3];
  const float* bk = (const float*)d_in[4];
  const float* Wv = (const float*)d_in[5];
  const float* bv = (const float*)d_in[6];
  const float* Wo = (const float*)d_in[7];
  const float* bo = (const float*)d_in[8];

  const int B = 4, T = SEQ, C = D_MODEL, M = B * T;
  __bf16* qb   = (__bf16*)d_ws;                 // 16 MB each
  __bf16* kb   = qb  + (size_t)M * C;
  __bf16* vtb  = kb  + (size_t)M * C;           // transposed [b][c][t]
  __bf16* attn = vtb + (size_t)M * C;

  dim3 gg(C / 128, M / 128);                    // (8, 64)
  wmma_gemm<0, 1><<<gg, 256, 0, stream>>>(x, Wq, bq, qb,  M, C, C);
  wmma_gemm<0, 1><<<gg, 256, 0, stream>>>(x, Wk, bk, kb,  M, C, C);
  wmma_gemm<0, 2><<<gg, 256, 0, stream>>>(x, Wv, bv, vtb, M, C, C);
  flash_attn<<<dim3(M / 16), 256, 0, stream>>>(qb, kb, vtb, attn);
  wmma_gemm<1, 0><<<gg, 256, 0, stream>>>(attn, Wo, bo, (float*)d_out, M, C, C);
}
